// RBACHierarchyGNN_10136122819017
// MI455X (gfx1250) — compile-verified, matrix operand order checked by reading, not run on hardware
//
#include <hip/hip_runtime.h>

// ---------------------------------------------------------------------------
// RBAC Hierarchy GNN for MI455X (gfx1250, wave32, WMMA bf16 + TDM/async-LDS)
// ---------------------------------------------------------------------------

typedef __attribute__((ext_vector_type(16))) __bf16 v16bf;
typedef __attribute__((ext_vector_type(8)))  __bf16 v8bf;
typedef __attribute__((ext_vector_type(8)))  float  v8f;
typedef unsigned int u32x4 __attribute__((ext_vector_type(4)));
typedef int          i32x4 __attribute__((ext_vector_type(4)));
typedef int          i32x8 __attribute__((ext_vector_type(8)));

#define N_NODES 50000
#define F_INPUT 256
#define HID     256
#define N_EDGES 400000
#define NLAYERS 3

#if __has_builtin(__builtin_amdgcn_tensor_load_to_lds)
#define HAVE_TDM 1
#else
#define HAVE_TDM 0
#endif

// ---------------------------------------------------------------------------
// Wait helpers
// ---------------------------------------------------------------------------
__device__ __forceinline__ void wait_async_lds() {
#if __has_builtin(__builtin_amdgcn_s_wait_asynccnt)
    __builtin_amdgcn_s_wait_asynccnt(0);
#else
    asm volatile("s_wait_asynccnt 0x0" ::: "memory");
#endif
}

__device__ __forceinline__ void wait_tensor() {
#if HAVE_TDM
    __builtin_amdgcn_s_wait_tensorcnt((short)0);
#endif
}

// ---------------------------------------------------------------------------
// Async global -> LDS, 16 bytes per lane (ASYNCcnt tracked)
// ---------------------------------------------------------------------------
__device__ __forceinline__ void async_b128_to_lds(void* ldsPtr, const void* gPtr) {
    unsigned ldsA = (unsigned)(unsigned long long)ldsPtr;   // low 32 bits = LDS offset
    unsigned long long ga = (unsigned long long)gPtr;
    asm volatile("global_load_async_to_lds_b128 %0, %1, off"
                 :: "v"(ldsA), "v"(ga) : "memory");
}

// ---------------------------------------------------------------------------
// TDM: load a [rows x 32] bf16 tile (row pitch = strideElems) into LDS.
// D# per CDNA5 ISA ch.8: group0 = {count, lds_addr, global_addr, type=2},
// group1 = {data_size=2B, tensor dims, tile dims, dim0 stride}. 2D tensor ->
// groups 2/3 zero. Issued by one wave (TDM ignores EXEC); TENSORcnt tracked.
// This toolchain's builtin takes 6 args (extra i32x8 group + cpol).
// ---------------------------------------------------------------------------
#if HAVE_TDM
__device__ __forceinline__ void tdm_load_tile_2d(void* ldsPtr, const void* gPtr,
                                                 int rows, int strideElems) {
    unsigned long long ga = (unsigned long long)gPtr;
    u32x4 g0;
    g0[0] = 1u;                                            // count=1 (valid user D#)
    g0[1] = (unsigned)(unsigned long long)ldsPtr;          // lds_addr (bytes)
    g0[2] = (unsigned)(ga & 0xffffffffu);                  // global_addr[31:0]
    g0[3] = (unsigned)((ga >> 32) & 0x01ffffffu) | (2u << 30); // addr[56:32] | type=2
    i32x8 g1;
    g1[0] = 0x00010000;                 // workgroup_mask=0, data_size=1 (2 bytes)
    g1[1] = (int)(32u << 16);           // tensor_dim0 = 32  (bits [79:48], low half)
    g1[2] = (int)((unsigned)rows << 16);// tensor_dim1 = rows (bits [111:80], low half)
    g1[3] = (int)(32u << 16);           // tile_dim0 = 32   (bits [127:112])
    g1[4] = rows & 0xffff;              // tile_dim1 = rows; tile_dim2 = 0
    g1[5] = strideElems;                // tensor_dim0_stride[31:0]
    g1[6] = 0;                          // stride[47:32]=0, dim1_stride low=0
    g1[7] = 0;
    i32x4 gz4; gz4[0] = 0; gz4[1] = 0; gz4[2] = 0; gz4[3] = 0;
    i32x8 gz8;
#pragma unroll
    for (int i = 0; i < 8; ++i) gz8[i] = 0;
    __builtin_amdgcn_tensor_load_to_lds(g0, g1, gz4, gz4, gz8, 0);
}
#endif

// ---------------------------------------------------------------------------
// LDS fragment loader. Tiles stored row-major [rows][32] bf16.
// bf16 16x32 A-fragment layout (B symmetric via W^T rows):
//   lanes 0-15 : row = lane,    elems 0-7 = K 0..7,   elems 8-15 = K 16..23
//   lanes 16-31: row = lane-16, elems 0-7 = K 8..15,  elems 8-15 = K 24..31
// ---------------------------------------------------------------------------
__device__ __forceinline__ v16bf ldsFrag(const __bf16* tile, int rowBase, int lane) {
    int r  = rowBase + (lane & 15);
    int ko = (lane >> 4) << 3;          // 0 or 8
    const __bf16* p = tile + r * 32 + ko;
    v8bf lo = *(const v8bf*)(p);
    v8bf hi = *(const v8bf*)(p + 16);
    v16bf out;
#pragma unroll
    for (int i = 0; i < 8; ++i) { out[i] = lo[i]; out[i + 8] = hi[i]; }
    return out;
}

__device__ __forceinline__ void zero_v8bf(__bf16* p) {
    v8bf z;
#pragma unroll
    for (int i = 0; i < 8; ++i) z[i] = (__bf16)0.f;
    *(v8bf*)p = z;
}

// ---------------------------------------------------------------------------
// Generic GEMM: C[M,N] = act(A[M,K] @ B[K,N] + bias), A bf16 row-major,
// BT = B^T bf16 row-major [N,K]. Block tile 64x128, 8 waves = 4(M) x 2(N),
// each wave accumulates a 16x64 strip via 4x v_wmma_f32_16x16x32_bf16.
// A staged via async-to-LDS; B^T tile staged via TDM.
// ---------------------------------------------------------------------------
__global__ void __launch_bounds__(256)
gemm_bf16_wmma(const __bf16* __restrict__ A, const __bf16* __restrict__ BT,
               const float* __restrict__ bias, float* __restrict__ outF,
               __bf16* __restrict__ outB, int M, int N, int K, int doRelu)
{
    __shared__ alignas(16) __bf16 sA[64 * 32];
    __shared__ alignas(16) __bf16 sB[128 * 32];

    const int tid   = threadIdx.x;
    const int lane  = tid & 31;
    const int wave  = tid >> 5;
    const int wm    = wave & 3;       // 0..3 : M group of 16
    const int wn    = wave >> 2;      // 0..1 : N group of 64
    const int mBase = blockIdx.x * 64;
    const int nBase = blockIdx.y * 128;

    v8f acc[4];
#pragma unroll
    for (int j = 0; j < 4; ++j)
#pragma unroll
        for (int i = 0; i < 8; ++i) acc[j][i] = 0.f;

    const int rowA  = tid >> 2;   // 0..63
    const int chunk = tid & 3;    // 0..3 (8 bf16 each)

    for (int k0 = 0; k0 < K; k0 += 32) {
        // ---- stage A (64 x 32): async global->LDS, zero-fill OOB rows ----
        {
            int gm = mBase + rowA;
            __bf16* lp = sA + rowA * 32 + chunk * 8;
            if (gm < M) {
                const __bf16* gp = A + (long)gm * K + k0 + chunk * 8;
                async_b128_to_lds(lp, gp);
                if (k0 + 32 < K) __builtin_prefetch(gp + 32, 0, 1);
            } else {
                zero_v8bf(lp);
            }
        }
        // ---- stage B^T (128 x 32) via Tensor Data Mover ----
#if HAVE_TDM
        if (tid == 0)
            tdm_load_tile_2d(sB, BT + (long)nBase * K + k0, 128, K);
#else
#pragma unroll
        for (int p = 0; p < 2; ++p) {
            int rb = rowA + p * 64;
            *(v8bf*)(sB + rb * 32 + chunk * 8) =
                *(const v8bf*)(BT + (long)(nBase + rb) * K + k0 + chunk * 8);
        }
#endif
        wait_async_lds();
        wait_tensor();
        __syncthreads();

        v16bf a = ldsFrag(sA, wm * 16, lane);
#pragma unroll
        for (int j = 0; j < 4; ++j) {
            v16bf b = ldsFrag(sB, wn * 64 + j * 16, lane);
            acc[j] = __builtin_amdgcn_wmma_f32_16x16x32_bf16(
                false, a, false, b, (short)0, acc[j], false, false);
        }
        __syncthreads();
    }

    // ---- epilogue: C frag elem r -> M = r + (lane>=16 ? 8 : 0), N = lane&15
#pragma unroll
    for (int j = 0; j < 4; ++j) {
        int nc = nBase + wn * 64 + j * 16 + (lane & 15);
        float bv = bias ? bias[nc] : 0.f;
#pragma unroll
        for (int r = 0; r < 8; ++r) {
            int mr = mBase + wm * 16 + r + ((lane >> 4) << 3);
            if (mr < M) {
                float v = acc[j][r] + bv;
                if (doRelu) v = fmaxf(v, 0.f);
                long idx = (long)mr * N + nc;
                if (outF) outF[idx] = v;
                if (outB) outB[idx] = (__bf16)v;
            }
        }
    }
}

// ---------------------------------------------------------------------------
// Fused edge MLP: hier[e,0:3] = relu(concat(h[src],h[dst]) @ Wh1 + bh1) @ Wh2 + bh2
// Block: 32 edges, full N=256, K=512.  8 waves = 2(M) x 4(N).
// A rows gathered on the fly (k<256 -> h[src], else h[dst]) via async-to-LDS.
// ---------------------------------------------------------------------------
__global__ void __launch_bounds__(256)
edge_mlp_wmma(const __bf16* __restrict__ hB,
              const int* __restrict__ src, const int* __restrict__ dst,
              const __bf16* __restrict__ Wh1T, const float* __restrict__ bh1,
              const float* __restrict__ Wh2,  const float* __restrict__ bh2,
              float* __restrict__ hier, int E)
{
    __shared__ alignas(16) __bf16 sA[32 * 32];
    __shared__ alignas(16) __bf16 sB[256 * 32];
    __shared__ float sH[32 * 256];

    const int tid   = threadIdx.x;
    const int lane  = tid & 31;
    const int wave  = tid >> 5;
    const int wm    = wave & 1;      // 0..1 : M group of 16
    const int wn    = wave >> 1;     // 0..3 : N group of 64
    const int mBase = blockIdx.x * 32;

    v8f acc[4];
#pragma unroll
    for (int j = 0; j < 4; ++j)
#pragma unroll
        for (int i = 0; i < 8; ++i) acc[j][i] = 0.f;

    const int chunk = tid & 3;
    for (int k0 = 0; k0 < 512; k0 += 32) {
        // ---- stage A (32 x 32), gathered via edge endpoints ----
        if (tid < 128) {
            int row = tid >> 2;                 // 0..31
            int e = mBase + row;
            __bf16* lp = sA + row * 32 + chunk * 8;
            if (e < E) {
                int kg = k0 + chunk * 8;
                int node = (kg < 256) ? src[e] : dst[e];
                async_b128_to_lds(lp, hB + (long)node * 256 + (kg & 255));
            } else {
                zero_v8bf(lp);
            }
        }
        // ---- stage Wh1^T (256 x 32) via TDM ----
#if HAVE_TDM
        if (tid == 0)
            tdm_load_tile_2d(sB, Wh1T + k0, 256, 512);
#else
        {
            int rowB = tid >> 2;
#pragma unroll
            for (int p = 0; p < 4; ++p) {
                int rb = rowB + p * 64;
                *(v8bf*)(sB + rb * 32 + chunk * 8) =
                    *(const v8bf*)(Wh1T + (long)rb * 512 + k0 + chunk * 8);
            }
        }
#endif
        wait_async_lds();
        wait_tensor();
        __syncthreads();

        v16bf a = ldsFrag(sA, wm * 16, lane);
#pragma unroll
        for (int j = 0; j < 4; ++j) {
            v16bf b = ldsFrag(sB, wn * 64 + j * 16, lane);
            acc[j] = __builtin_amdgcn_wmma_f32_16x16x32_bf16(
                false, a, false, b, (short)0, acc[j], false, false);
        }
        __syncthreads();
    }

    // ---- relu hidden into LDS ----
#pragma unroll
    for (int j = 0; j < 4; ++j) {
        int nl = wn * 64 + j * 16 + (lane & 15);
        float bv = bh1[nl];
#pragma unroll
        for (int r = 0; r < 8; ++r) {
            int ml = wm * 16 + r + ((lane >> 4) << 3);
            sH[ml * 256 + nl] = fmaxf(acc[j][r] + bv, 0.f);
        }
    }
    __syncthreads();

    // ---- fused 256 -> 3 projection ----
    if (tid < 96) {
        int el = tid / 3, jj = tid % 3;
        float s = bh2[jj];
        for (int n2 = 0; n2 < 256; ++n2) s += sH[el * 256 + n2] * Wh2[n2 * 3 + jj];
        int e = mBase + el;
        if (e < E) hier[(long)e * 3 + jj] = s;
    }
}

// ---------------------------------------------------------------------------
// Fused permission head: perm[n] = sigmoid(relu(h @ Wp1 + bp1) @ Wp2 + bp2)
// Block: 64 nodes, N=128, K=256.  8 waves = 4(M) x 2(N).
// ---------------------------------------------------------------------------
__global__ void __launch_bounds__(256)
perm_mlp_wmma(const __bf16* __restrict__ hB, const __bf16* __restrict__ Wp1T,
              const float* __restrict__ bp1, const float* __restrict__ Wp2,
              const float* __restrict__ bp2, float* __restrict__ perm, int M)
{
    __shared__ alignas(16) __bf16 sA[64 * 32];
    __shared__ alignas(16) __bf16 sB[128 * 32];
    __shared__ float sH[64 * 128];

    const int tid   = threadIdx.x;
    const int lane  = tid & 31;
    const int wave  = tid >> 5;
    const int wm    = wave & 3;
    const int wn    = wave >> 2;
    const int mBase = blockIdx.x * 64;

    v8f acc[4];
#pragma unroll
    for (int j = 0; j < 4; ++j)
#pragma unroll
        for (int i = 0; i < 8; ++i) acc[j][i] = 0.f;

    const int rowA  = tid >> 2;
    const int chunk = tid & 3;
    for (int k0 = 0; k0 < 256; k0 += 32) {
        {
            int gm = mBase + rowA;
            __bf16* lp = sA + rowA * 32 + chunk * 8;
            if (gm < M)
                async_b128_to_lds(lp, hB + (long)gm * 256 + k0 + chunk * 8);
            else
                zero_v8bf(lp);
        }
#if HAVE_TDM
        if (tid == 0)
            tdm_load_tile_2d(sB, Wp1T + k0, 128, 256);
#else
#pragma unroll
        for (int p = 0; p < 2; ++p) {
            int rb = rowA + p * 64;
            *(v8bf*)(sB + rb * 32 + chunk * 8) =
                *(const v8bf*)(Wp1T + (long)rb * 256 + k0 + chunk * 8);
        }
#endif
        wait_async_lds();
        wait_tensor();
        __syncthreads();

        v16bf a = ldsFrag(sA, wm * 16, lane);
#pragma unroll
        for (int j = 0; j < 4; ++j) {
            v16bf b = ldsFrag(sB, wn * 64 + j * 16, lane);
            acc[j] = __builtin_amdgcn_wmma_f32_16x16x32_bf16(
                false, a, false, b, (short)0, acc[j], false, false);
        }
        __syncthreads();
    }

#pragma unroll
    for (int j = 0; j < 4; ++j) {
        int nl = wn * 64 + j * 16 + (lane & 15);
        float bv = bp1[nl];
#pragma unroll
        for (int r = 0; r < 8; ++r) {
            int ml = wm * 16 + r + ((lane >> 4) << 3);
            sH[ml * 128 + nl] = fmaxf(acc[j][r] + bv, 0.f);
        }
    }
    __syncthreads();

    if (tid < 64) {
        float s = bp2[0];
        for (int n2 = 0; n2 < 128; ++n2) s += sH[tid * 128 + n2] * Wp2[n2];
        int m = mBase + tid;
        if (m < M) perm[m] = 1.f / (1.f + __expf(-s));
    }
}

// ---------------------------------------------------------------------------
// Elementwise / graph kernels
// ---------------------------------------------------------------------------
__global__ void f32_to_bf16_k(const float* __restrict__ x, __bf16* __restrict__ y, long n) {
    long t = (long)blockIdx.x * blockDim.x + threadIdx.x;
    if (t < n) y[t] = (__bf16)x[t];
}

// W [K,N] fp32 row-major -> WT [N,K] bf16 row-major
__global__ void transpose_bf16_k(const float* __restrict__ W, __bf16* __restrict__ WT,
                                 int K, int N) {
    long t = (long)blockIdx.x * blockDim.x + threadIdx.x;
    if (t < (long)K * N) {
        int k = (int)(t / N), n = (int)(t % N);
        WT[(long)n * K + k] = (__bf16)W[(long)k * N + n];
    }
}

__global__ void deg_init_k(float* __restrict__ deg, int n) {
    int t = blockIdx.x * blockDim.x + threadIdx.x;
    if (t < n) deg[t] = 1.f;   // self loop
}

__global__ void deg_edges_k(const int* __restrict__ dst, float* __restrict__ deg, int E) {
    int t = blockIdx.x * blockDim.x + threadIdx.x;
    if (t < E) atomicAdd(&deg[dst[t]], 1.f);
}

__global__ void norm_k(const float* __restrict__ deg, float* __restrict__ nrm, int n) {
    int t = blockIdx.x * blockDim.x + threadIdx.x;
    if (t < n) nrm[t] = rsqrtf(fmaxf(deg[t], 1.f));
}

// agg[i,f] = bg[f] + norm[i]^2 * hw[i,f]   (bias + self-loop contribution)
__global__ void agg_init_k(const float* __restrict__ hw, const float* __restrict__ nrm,
                           const float* __restrict__ bg, float* __restrict__ agg, long total) {
    long t = (long)blockIdx.x * blockDim.x + threadIdx.x;
    if (t < total) {
        int i = (int)(t >> 8);       // /256
        int f = (int)(t & 255);
        float nv = nrm[i];
        agg[t] = bg[f] + nv * nv * hw[t];
    }
}

// one block (256 threads) per edge: agg[dst, f] += norm[s]*norm[d]*hw[src, f]
__global__ void gcn_scatter_k(const float* __restrict__ hw, const int* __restrict__ src,
                              const int* __restrict__ dst, const float* __restrict__ nrm,
                              float* __restrict__ agg, int E) {
    int e = blockIdx.x;
    if (e >= E) return;
    int s = src[e], d = dst[e];
    float c = nrm[s] * nrm[d];
    int f = threadIdx.x;
    atomicAdd(&agg[(long)d * 256 + f], c * hw[(long)s * 256 + f]);
}

// h = relu(agg); write fp32 output slot(s) and bf16 mirror
__global__ void gcn_final_k(const float* __restrict__ agg, float* __restrict__ interL,
                            float* __restrict__ hOut, __bf16* __restrict__ hB, long total) {
    long t = (long)blockIdx.x * blockDim.x + threadIdx.x;
    if (t < total) {
        float v = fmaxf(agg[t], 0.f);
        interL[t] = v;
        hB[t] = (__bf16)v;
        if (hOut) hOut[t] = v;
    }
}

// ---------------------------------------------------------------------------
// Launcher
// ---------------------------------------------------------------------------
extern "C" void kernel_launch(void* const* d_in, const int* in_sizes, int n_in,
                              void* d_out, int out_size, void* d_ws, size_t ws_size,
                              hipStream_t stream) {
    (void)in_sizes; (void)n_in; (void)out_size; (void)ws_size;

    const int N = N_NODES, FIN = F_INPUT, H = HID, E = N_EDGES;

    const float* x    = (const float*)d_in[0];
    const int*   ei   = (const int*)d_in[1];
    const int*   src  = ei;
    const int*   dst  = ei + E;
    const float* W_in = (const float*)d_in[2];
    const float* b_in = (const float*)d_in[3];
    const float* Wg   = (const float*)d_in[4];   // [L,H,H]
    const float* bg   = (const float*)d_in[5];   // [L,H]
    const float* Wh1  = (const float*)d_in[6];   // [2H,H]
    const float* bh1  = (const float*)d_in[7];
    const float* Wh2  = (const float*)d_in[8];   // [H,3]
    const float* bh2  = (const float*)d_in[9];
    const float* Wp1  = (const float*)d_in[10];  // [H,H/2]
    const float* bp1  = (const float*)d_in[11];
    const float* Wp2  = (const float*)d_in[12];  // [H/2,1]
    const float* bp2  = (const float*)d_in[13];

    float* out    = (float*)d_out;
    float* h_out  = out;                         // [N,H]
    float* hier   = out + (size_t)N * H;         // [E,3]
    float* perm   = hier + (size_t)E * 3;        // [N,1]
    float* inter  = perm + N;                    // [L,N,H]

    // ---- workspace carve-up ----
    char* ws = (char*)d_ws;
    size_t off = 0;
    auto carve = [&](size_t bytes) -> char* {
        char* p = ws + off;
        off = (off + bytes + 255) & ~(size_t)255;
        return p;
    };
    __bf16* xB    = (__bf16*)carve((size_t)N * FIN * 2);
    __bf16* hB    = (__bf16*)carve((size_t)N * H * 2);
    float*  hw    = (float*) carve((size_t)N * H * 4);
    float*  agg   = (float*) carve((size_t)N * H * 4);
    float*  deg   = (float*) carve((size_t)N * 4);
    float*  nrm   = (float*) carve((size_t)N * 4);
    __bf16* WinT  = (__bf16*)carve((size_t)FIN * H * 2);
    __bf16* WgT   = (__bf16*)carve((size_t)NLAYERS * H * H * 2);
    __bf16* Wh1T  = (__bf16*)carve((size_t)2 * H * H * 2);
    __bf16* Wp1T  = (__bf16*)carve((size_t)H * (H / 2) * 2);

    const int TB = 256;
    auto blk1 = [](long n) { return (unsigned)((n + 255) / 256); };

    // ---- 0) precompute: bf16 copies / transposed weights / norms ----
    f32_to_bf16_k<<<blk1((long)N * FIN), TB, 0, stream>>>(x, xB, (long)N * FIN);
    transpose_bf16_k<<<blk1((long)FIN * H), TB, 0, stream>>>(W_in, WinT, FIN, H);
    for (int l = 0; l < NLAYERS; ++l)
        transpose_bf16_k<<<blk1((long)H * H), TB, 0, stream>>>(
            Wg + (size_t)l * H * H, WgT + (size_t)l * H * H, H, H);
    transpose_bf16_k<<<blk1((long)2 * H * H), TB, 0, stream>>>(Wh1, Wh1T, 2 * H, H);
    transpose_bf16_k<<<blk1((long)H * (H / 2)), TB, 0, stream>>>(Wp1, Wp1T, H, H / 2);

    deg_init_k<<<blk1(N), TB, 0, stream>>>(deg, N);
    deg_edges_k<<<blk1(E), TB, 0, stream>>>(dst, deg, E);
    norm_k<<<blk1(N), TB, 0, stream>>>(deg, nrm, N);

    // ---- 1) input layer: hB = bf16(relu(x @ W_in + b_in)) ----
    {
        dim3 grid((N + 63) / 64, H / 128);
        gemm_bf16_wmma<<<grid, TB, 0, stream>>>(xB, WinT, b_in, nullptr, hB,
                                                N, H, FIN, 1);
    }

    // ---- 2) GCN layers ----
    const long totNH = (long)N * H;
    for (int l = 0; l < NLAYERS; ++l) {
        dim3 grid((N + 63) / 64, H / 128);
        gemm_bf16_wmma<<<grid, TB, 0, stream>>>(hB, WgT + (size_t)l * H * H,
                                                nullptr, hw, nullptr, N, H, H, 0);
        agg_init_k<<<blk1(totNH), TB, 0, stream>>>(hw, nrm, bg + (size_t)l * H, agg, totNH);
        gcn_scatter_k<<<E, TB, 0, stream>>>(hw, src, dst, nrm, agg, E);
        float* interL = inter + (size_t)l * N * H;
        float* hO = (l == NLAYERS - 1) ? h_out : nullptr;
        gcn_final_k<<<blk1(totNH), TB, 0, stream>>>(agg, interL, hO, hB, totNH);
    }

    // ---- 3) fused edge MLP -> hier ----
    edge_mlp_wmma<<<(E + 31) / 32, TB, 0, stream>>>(hB, src, dst, Wh1T, bh1,
                                                    Wh2, bh2, hier, E);

    // ---- 4) fused permission head -> perm ----
    perm_mlp_wmma<<<(N + 63) / 64, TB, 0, stream>>>(hB, Wp1T, bp1, Wp2, bp2, perm, N);
}
